// Model_34162169872880
// MI455X (gfx1250) — matrix operand order repaired and checked by reading, not compiled, not run on hardware
//
#include <hip/hip_runtime.h>

// SRU++ LM forward (L=1024,B=4,D=1024,PROJ=256,H=8,HD=32,DEPTH=4,V=32000)
// All matmuls via v_wmma_f32_16x16x32_bf16 (bf16 inputs, f32 accumulate).
// GEMM operands are pre-converted to bf16 once (A flat, B transposed to [N][K])
// so the GEMM hot loop is pure b128 copies + ds_load_b128 fragment loads + WMMA.

#define L_    1024
#define B_    4
#define D_    1024
#define PROJ_ 256
#define H_    8
#define HD_   32
#define DEPTH_ 4
#define V_    32000
#define LB_   (L_*B_)          // 4096 rows
#define NEGMASK (-10000.0f)
#define VC_   3200             // vocab chunk (32000 = 10*3200, 3200 = 25*128)

typedef __attribute__((ext_vector_type(16))) __bf16 bf16x16;
typedef __attribute__((ext_vector_type(8)))  float  f32x8;

__device__ __forceinline__ unsigned short f2bf(float f) {
  unsigned u = __builtin_bit_cast(unsigned, f);
  u += 0x7FFFu + ((u >> 16) & 1u);           // round-to-nearest-even
  return (unsigned short)(u >> 16);
}
__device__ __forceinline__ __bf16 bfbits(unsigned short s) {
  return __builtin_bit_cast(__bf16, s);
}
__device__ __forceinline__ float sigmoidf_(float x) {
  return 1.0f / (1.0f + __expf(-x));
}

// ---------------------------------------------------------------------------
// Flat f32 -> bf16 conversion (A operands). n must be a multiple of 1024.
// ---------------------------------------------------------------------------
__global__ __launch_bounds__(256)
void cvt_a_kernel(const float* __restrict__ in, unsigned short* __restrict__ out)
{
  int i = blockIdx.x * 256 + threadIdx.x;    // index in float4 units
  float4 v = ((const float4*)in)[i];
  ushort4 w;
  w.x = f2bf(v.x); w.y = f2bf(v.y); w.z = f2bf(v.z); w.w = f2bf(v.w);
  ((ushort4*)out)[i] = w;
}

// ---------------------------------------------------------------------------
// f32 [K][ldin] (cols c0..c0+N-1) -> bf16 transposed [N][K].
// 32x32 tile transpose through padded LDS; both global sides coalesced.
// Grid: (N/32, K/32), 256 threads.
// ---------------------------------------------------------------------------
__global__ __launch_bounds__(256)
void cvt_bt_kernel(const float* __restrict__ in, unsigned short* __restrict__ out,
                   int K, int ldin, int c0)
{
  __shared__ unsigned short t[32][33];
  const int nt = blockIdx.x * 32, kt = blockIdx.y * 32;
  const int tx = threadIdx.x & 31, ty = threadIdx.x >> 5;   // ty: 0..7
  #pragma unroll
  for (int i = 0; i < 4; ++i) {
    int k = kt + ty + i * 8;
    t[tx][ty + i * 8] = f2bf(in[(size_t)k * ldin + c0 + nt + tx]);
  }
  __syncthreads();
  #pragma unroll
  for (int i = 0; i < 4; ++i) {
    int n = nt + ty + i * 8;
    out[(size_t)n * K + kt + tx] = t[ty + i * 8][tx];
  }
}

// ---------------------------------------------------------------------------
// GEMM: C[M,N] = A[M,K] @ B[K,N] (+ bias[N]); A is bf16 [M][K], B is bf16
// pre-transposed BT [N][K]. Block tile 128x128x32, 8 waves (4Mx2N), wave tile
// 32x64 = 2x4 WMMA accumulators. Staging = pure uint4 copies; both fragment
// layouts are contiguous 16-value K-runs -> ds_load_b128 pairs.
// Requires M%128==0, N%128==0, K%32==0 (true at all call sites).
// ---------------------------------------------------------------------------
__global__ __launch_bounds__(256)
void gemm_bf16(const unsigned short* __restrict__ A,
               const unsigned short* __restrict__ BT,
               const float* __restrict__ bias, float* __restrict__ C,
               int K, int ldc)
{
  __shared__ __align__(16) unsigned short As[128 * 32];   // [m][k]
  __shared__ __align__(16) unsigned short Bs[128 * 32];   // [n][k]

  const int tid  = threadIdx.x;
  const int lane = tid & 31;
  const int wid  = tid >> 5;
  const int wm   = (wid & 3) * 32;          // wave M offset within tile
  const int wn   = (wid >> 2) * 64;         // wave N offset within tile
  const int bm   = blockIdx.y * 128;
  const int bn   = blockIdx.x * 128;

  f32x8 acc[2][4];
  #pragma unroll
  for (int i = 0; i < 2; ++i)
    #pragma unroll
    for (int j = 0; j < 4; ++j) acc[i][j] = f32x8{};

  const int rowA = lane & 15;               // A frag: M row
  const int kbA  = (lane < 16) ? 0 : 16;    // A frag: K base (contiguous K order, same as B)
  const int colB = lane & 15;               // B frag: N col
  const int kbB  = (lane < 16) ? 0 : 16;    // B frag: K base

  for (int kt = 0; kt < K; kt += 32) {
    if (kt + 32 < K) {                      // prefetch next K tiles
      __builtin_prefetch(&A[(size_t)(bm + (tid >> 1)) * K + kt + 32], 0, 0);
      __builtin_prefetch(&BT[(size_t)(bn + (tid >> 1)) * K + kt + 32], 0, 0);
    }
    // Stage both 128x32 bf16 tiles: 2x uint4 (8 bf16) per thread per side.
    #pragma unroll
    for (int t = 0; t < 2; ++t) {
      int id = t * 256 + tid;
      int r = id >> 2, c8 = (id & 3) * 8;
      *(uint4*)&As[r * 32 + c8] = *(const uint4*)(A  + (size_t)(bm + r) * K + kt + c8);
      *(uint4*)&Bs[r * 32 + c8] = *(const uint4*)(BT + (size_t)(bn + r) * K + kt + c8);
    }
    __syncthreads();

    bf16x16 af[2];
    #pragma unroll
    for (int i = 0; i < 2; ++i) {
      int r = wm + i * 16 + rowA;
      #pragma unroll
      for (int e = 0; e < 16; ++e) {
        int k = kbA + e;                          // K = {kb..kb+15}
        af[i][e] = bfbits(As[r * 32 + k]);
      }
    }
    bf16x16 bq[4];
    #pragma unroll
    for (int j = 0; j < 4; ++j) {
      int cn = wn + j * 16 + colB;
      #pragma unroll
      for (int e = 0; e < 16; ++e)                // contiguous -> ds_load_b128
        bq[j][e] = bfbits(Bs[cn * 32 + kbB + e]);
    }
    #pragma unroll
    for (int i = 0; i < 2; ++i)
      #pragma unroll
      for (int j = 0; j < 4; ++j)
        acc[i][j] = __builtin_amdgcn_wmma_f32_16x16x32_bf16(
            false, af[i], false, bq[j], (short)0, acc[i][j], false, false);
    __syncthreads();
  }

  const int rhalf = (lane >> 4) * 8;        // C/D layout: M = r (+8 for hi lanes)
  #pragma unroll
  for (int i = 0; i < 2; ++i)
    #pragma unroll
    for (int j = 0; j < 4; ++j) {
      int cn = bn + wn + j * 16 + (lane & 15);
      float bv = bias ? bias[cn] : 0.0f;
      #pragma unroll
      for (int r = 0; r < 8; ++r) {
        int rm = bm + wm + i * 16 + rhalf + r;
        C[(size_t)rm * ldc + cn] = acc[i][j][r] + bv;
      }
    }
}

// ---------------------------------------------------------------------------
// Causal flash attention per (b, head, 16-query tile), single wave per block.
// q = z-slice * HD^-0.5 ; scores via WMMA(q, k^T) + additive -1e4 mask ;
// online softmax ; out accumulated via WMMA(probs, v) ; epilogue applies the
// rezero residual: att = (out / l) * alpha + z.
// ---------------------------------------------------------------------------
__global__ __launch_bounds__(32)
void attn_kernel(const float* __restrict__ z, const float* __restrict__ kv,
                 const float* __restrict__ alpha_p, int layer,
                 float* __restrict__ att)
{
  __shared__ unsigned short Ps[16 * 32];    // prob tile relayout (C -> A frag)
  const int lane = threadIdx.x;
  const int tile = blockIdx.x;              // 0..63 (query tile)
  const int b    = blockIdx.y;
  const int hh   = blockIdx.z;
  const float alpha = alpha_p[layer];
  const float qs = 0.17677669529663687f;    // 32^-0.5

  const int rowA = lane & 15;
  const int kbA  = (lane < 16) ? 0 : 16;
  const int colB = lane & 15;
  const int kbB  = (lane < 16) ? 0 : 16;
  const int rhalf = (lane >> 4) * 8;
  const int ql0  = tile * 16;

  // A fragment: q tile, 16 queries x HD=32. Per lane: one contiguous 16-float
  // run (kbA..kbA+15) -> float4 loads.
  bf16x16 qf;
  {
    const float* qp = z + (size_t)((ql0 + rowA) * B_ + b) * PROJ_ + hh * HD_;
    #pragma unroll
    for (int q4 = 0; q4 < 2; ++q4) {
      float4 v0 = *(const float4*)(qp + kbA + q4 * 4);
      float4 v1 = *(const float4*)(qp + kbA + 8 + q4 * 4);
      qf[q4*4+0] = bfbits(f2bf(v0.x * qs)); qf[q4*4+1] = bfbits(f2bf(v0.y * qs));
      qf[q4*4+2] = bfbits(f2bf(v0.z * qs)); qf[q4*4+3] = bfbits(f2bf(v0.w * qs));
      qf[8+q4*4+0] = bfbits(f2bf(v1.x * qs)); qf[8+q4*4+1] = bfbits(f2bf(v1.y * qs));
      qf[8+q4*4+2] = bfbits(f2bf(v1.z * qs)); qf[8+q4*4+3] = bfbits(f2bf(v1.w * qs));
    }
  }

  f32x8 o0 = f32x8{}, o1 = f32x8{};
  float mrow[8], lrow[8];
  #pragma unroll
  for (int r = 0; r < 8; ++r) { mrow[r] = -1e30f; lrow[r] = 0.0f; }

  const int nchunks = tile / 2 + 1;         // covers keys 0 .. tile*16+15
  for (int cb = 0; cb < nchunks; ++cb) {
    const int base = cb * 32;
    // B fragments: k^T (K=HD dims contiguous per lane -> float4 loads)
    bf16x16 k0, k1;
    {
      const float* kp0 = kv + (size_t)((base + colB) * B_ + b) * (2 * PROJ_) + hh * HD_ + kbB;
      const float* kp1 = kv + (size_t)((base + 16 + colB) * B_ + b) * (2 * PROJ_) + hh * HD_ + kbB;
      #pragma unroll
      for (int q4 = 0; q4 < 4; ++q4) {
        float4 a = *(const float4*)(kp0 + q4 * 4);
        float4 c = *(const float4*)(kp1 + q4 * 4);
        k0[q4*4+0] = bfbits(f2bf(a.x)); k0[q4*4+1] = bfbits(f2bf(a.y));
        k0[q4*4+2] = bfbits(f2bf(a.z)); k0[q4*4+3] = bfbits(f2bf(a.w));
        k1[q4*4+0] = bfbits(f2bf(c.x)); k1[q4*4+1] = bfbits(f2bf(c.y));
        k1[q4*4+2] = bfbits(f2bf(c.z)); k1[q4*4+3] = bfbits(f2bf(c.w));
      }
    }
    f32x8 zc = f32x8{};
    f32x8 s0 = __builtin_amdgcn_wmma_f32_16x16x32_bf16(false, qf, false, k0, (short)0, zc, false, false);
    f32x8 s1 = __builtin_amdgcn_wmma_f32_16x16x32_bf16(false, qf, false, k1, (short)0, zc, false, false);

    #pragma unroll
    for (int r = 0; r < 8; ++r) {
      int ql = ql0 + rhalf + r;
      float a0 = s0[r] + (((base + colB) > ql) ? NEGMASK : 0.0f);
      float a1 = s1[r] + (((base + 16 + colB) > ql) ? NEGMASK : 0.0f);
      float cm = fmaxf(a0, a1);
      #pragma unroll
      for (int off = 1; off < 16; off <<= 1) cm = fmaxf(cm, __shfl_xor(cm, off, 32));
      float mn = fmaxf(mrow[r], cm);
      float scale = __expf(mrow[r] - mn);
      float p0 = __expf(a0 - mn);
      float p1 = __expf(a1 - mn);
      float rs = p0 + p1;
      #pragma unroll
      for (int off = 1; off < 16; off <<= 1) rs += __shfl_xor(rs, off, 32);
      lrow[r] = lrow[r] * scale + rs;
      mrow[r] = mn;
      o0[r] *= scale; o1[r] *= scale;
      int row = rhalf + r;
      Ps[row * 32 + colB]      = f2bf(p0);  // DS ops in-order within a wave
      Ps[row * 32 + 16 + colB] = f2bf(p1);
    }
    // A fragment: prob tile (16 queries x 32 chunk-keys)
    bf16x16 pf;
    #pragma unroll
    for (int e = 0; e < 16; ++e) {
      int kk = kbA + e;
      pf[e] = bfbits(Ps[rowA * 32 + kk]);
    }
    // B fragments: v (K=32 chunk keys, N=HD dims in two 16-col halves)
    bf16x16 v0, v1;
    #pragma unroll
    for (int e = 0; e < 16; ++e) {
      int kk = kbB + e;
      size_t vb = (size_t)((base + kk) * B_ + b) * (2 * PROJ_) + PROJ_ + hh * HD_;
      v0[e] = bfbits(f2bf(kv[vb + colB]));
      v1[e] = bfbits(f2bf(kv[vb + 16 + colB]));
    }
    o0 = __builtin_amdgcn_wmma_f32_16x16x32_bf16(false, pf, false, v0, (short)0, o0, false, false);
    o1 = __builtin_amdgcn_wmma_f32_16x16x32_bf16(false, pf, false, v1, (short)0, o1, false, false);
  }

  #pragma unroll
  for (int r = 0; r < 8; ++r) {
    int ql = ql0 + rhalf + r;
    float inv = 1.0f / lrow[r];
    size_t idx = (size_t)(ql * B_ + b) * PROJ_ + hh * HD_;
    att[idx + colB]      = o0[r] * inv * alpha + z[idx + colB];
    att[idx + 16 + colB] = o1[r] * inv * alpha + z[idx + 16 + colB];
  }
}

// ---------------------------------------------------------------------------
// SRU elementwise highway recurrence: serial in L, parallel over B*D.
// U layout (L,B,3D) with (d,j) -> 3*d+j interleave (reshape (L,B,D,3)).
// ---------------------------------------------------------------------------
__global__ __launch_bounds__(256)
void sru_kernel(const float* __restrict__ U, const float* __restrict__ hin,
                const float* __restrict__ c0p, const float* __restrict__ wc,
                const float* __restrict__ bias, float* __restrict__ hout)
{
  int idx = blockIdx.x * 256 + threadIdx.x;   // 0..B*D-1
  int b = idx / D_, d = idx % D_;
  float c  = c0p[idx];
  float vf = wc[d],   vr = wc[D_ + d];
  float bf = bias[d], br = bias[D_ + d];
  for (int l = 0; l < L_; ++l) {
    size_t ub = ((size_t)(l * B_ + b) * D_ + d) * 3;
    float u0 = U[ub], u1 = U[ub + 1] + bf, u2 = U[ub + 2] + br;
    float f = sigmoidf_(u1 + vf * c);
    c = f * c + (1.0f - f) * u0;
    float r = sigmoidf_(u2 + vr * c);
    size_t hi = (size_t)(l * B_ + b) * D_ + d;
    hout[hi] = r * c + (1.0f - r) * hin[hi];
  }
}

__global__ __launch_bounds__(256)
void embed_kernel(const int* __restrict__ x, const float* __restrict__ embW,
                  float* __restrict__ h)
{
  size_t i = (size_t)blockIdx.x * 256 + threadIdx.x;
  int lb = (int)(i / D_), d = (int)(i % D_);
  h[i] = embW[(size_t)x[lb] * D_ + d];
}

// Per-row max/sum-exp over one vocab chunk, merged online into (sm, ss);
// captures target logit when y falls in this chunk.
__global__ __launch_bounds__(256)
void rowstat_kernel(const float* __restrict__ chunk, int vc, int c0,
                    const int* __restrict__ y, float* __restrict__ sm,
                    float* __restrict__ ss, float* __restrict__ st, int first)
{
  __shared__ float red[256];
  const int row = blockIdx.x;
  const int tid = threadIdx.x;
  const float* p = chunk + (size_t)row * vc;
  float vmax = -1e30f;
  for (int j = tid; j < vc; j += 256) vmax = fmaxf(vmax, p[j]);
  red[tid] = vmax; __syncthreads();
  for (int s = 128; s > 0; s >>= 1) {
    if (tid < s) red[tid] = fmaxf(red[tid], red[tid + s]);
    __syncthreads();
  }
  float rm = red[0]; __syncthreads();
  float vsum = 0.0f;
  for (int j = tid; j < vc; j += 256) vsum += __expf(p[j] - rm);
  red[tid] = vsum; __syncthreads();
  for (int s = 128; s > 0; s >>= 1) {
    if (tid < s) red[tid] += red[tid + s];
    __syncthreads();
  }
  if (tid == 0) {
    float rs = red[0];
    int yt = y[row];
    if (yt >= c0 && yt < c0 + vc) st[row] = p[yt - c0];
    if (first) { sm[row] = rm; ss[row] = rs; }
    else {
      float m0 = sm[row];
      float mn = fmaxf(m0, rm);
      ss[row] = ss[row] * __expf(m0 - mn) + rs * __expf(rm - mn);
      sm[row] = mn;
    }
  }
}

__global__ __launch_bounds__(256)
void loss_kernel(const float* __restrict__ sm, const float* __restrict__ ss,
                 const float* __restrict__ st, float* __restrict__ out)
{
  int row = blockIdx.x * 256 + threadIdx.x;
  out[row] = sm[row] + __logf(ss[row]) - st[row];   // LSE - target logit
}

// ---------------------------------------------------------------------------
extern "C" void kernel_launch(void* const* d_in, const int* in_sizes, int n_in,
                              void* d_out, int out_size, void* d_ws, size_t ws_size,
                              hipStream_t stream)
{
  (void)in_sizes; (void)n_in; (void)out_size; (void)ws_size;
  const int*   x      = (const int*)  d_in[0];
  const int*   y      = (const int*)  d_in[1];
  const float* hidden = (const float*)d_in[2];
  const float* embW   = (const float*)d_in[3];
  const float* W1     = (const float*)d_in[4];
  const float* W2     = (const float*)d_in[5];
  const float* W3     = (const float*)d_in[6];
  const float* alpha  = (const float*)d_in[7];
  const float* wc     = (const float*)d_in[8];
  const float* bias   = (const float*)d_in[9];
  const float* Wout   = (const float*)d_in[10];
  const float* bout   = (const float*)d_in[11];

  char* ws = (char*)d_ws;
  size_t off = 0;
  auto alloc = [&](size_t bytes) -> void* {
    void* p = (void*)(ws + off);
    off += (bytes + 255) & ~(size_t)255;
    return p;
  };
  float* h0   = (float*)alloc((size_t)LB_ * D_ * 4);          // 16.8 MB
  float* h1   = (float*)alloc((size_t)LB_ * D_ * 4);          // 16.8 MB
  float* z    = (float*)alloc((size_t)LB_ * PROJ_ * 4);       //  4.2 MB
  float* kvb  = (float*)alloc((size_t)LB_ * 2 * PROJ_ * 4);   //  8.4 MB
  float* att  = (float*)alloc((size_t)LB_ * PROJ_ * 4);       //  4.2 MB
  float* Ubuf = (float*)alloc((size_t)LB_ * VC_ * 4);         // 52.4 MB (U / logits chunk)
  unsigned short* Abf = (unsigned short*)alloc((size_t)LB_ * D_ * 2);   // 8.4 MB
  unsigned short* Wbf = (unsigned short*)alloc((size_t)VC_ * D_ * 2);   // 6.6 MB
  float* sm   = (float*)alloc((size_t)LB_ * 4);
  float* ss   = (float*)alloc((size_t)LB_ * 4);
  float* st   = (float*)alloc((size_t)LB_ * 4);

  embed_kernel<<<dim3(LB_ * D_ / 256), 256, 0, stream>>>(x, embW, h0);

  float* hin = h0; float* hout = h1;
  for (int i = 0; i < DEPTH_; ++i) {
    // z = h @ W1[i]           (4096 x 256  x 1024)
    cvt_a_kernel<<<dim3(LB_ * D_ / 4 / 256), 256, 0, stream>>>(hin, Abf);
    cvt_bt_kernel<<<dim3(PROJ_ / 32, D_ / 32), 256, 0, stream>>>(
        W1 + (size_t)i * D_ * PROJ_, Wbf, D_, PROJ_, 0);
    gemm_bf16<<<dim3(PROJ_ / 128, LB_ / 128), 256, 0, stream>>>(
        Abf, Wbf, nullptr, z, D_, PROJ_);
    // kv = z @ W2[i]          (4096 x 512  x 256)
    cvt_a_kernel<<<dim3(LB_ * PROJ_ / 4 / 256), 256, 0, stream>>>(z, Abf);
    cvt_bt_kernel<<<dim3(2 * PROJ_ / 32, PROJ_ / 32), 256, 0, stream>>>(
        W2 + (size_t)i * PROJ_ * 2 * PROJ_, Wbf, PROJ_, 2 * PROJ_, 0);
    gemm_bf16<<<dim3(2 * PROJ_ / 128, LB_ / 128), 256, 0, stream>>>(
        Abf, Wbf, nullptr, kvb, PROJ_, 2 * PROJ_);
    // att = softmax(qk^T+mask) v * alpha + z   (flash, fused rezero)
    attn_kernel<<<dim3(L_ / 16, B_, H_), 32, 0, stream>>>(z, kvb, alpha, i, att);
    // U = att @ W3[i]         (4096 x 3072 x 256)
    cvt_a_kernel<<<dim3(LB_ * PROJ_ / 4 / 256), 256, 0, stream>>>(att, Abf);
    cvt_bt_kernel<<<dim3(3 * D_ / 32, PROJ_ / 32), 256, 0, stream>>>(
        W3 + (size_t)i * PROJ_ * 3 * D_, Wbf, PROJ_, 3 * D_, 0);
    gemm_bf16<<<dim3(3 * D_ / 128, LB_ / 128), 256, 0, stream>>>(
        Abf, Wbf, nullptr, Ubuf, PROJ_, 3 * D_);
    // SRU recurrence (serial in L, parallel over B*D)
    sru_kernel<<<dim3(B_ * D_ / 256), 256, 0, stream>>>(
        Ubuf, hin, hidden + (size_t)i * B_ * D_,
        wc + (size_t)i * 2 * D_, bias + (size_t)i * 2 * D_, hout);
    float* t = hin; hin = hout; hout = t;
  }

  // logits = h @ Wout + bout, chunked over V with online log-sum-exp.
  cvt_a_kernel<<<dim3(LB_ * D_ / 4 / 256), 256, 0, stream>>>(hin, Abf);
  for (int c = 0; c < V_ / VC_; ++c) {
    cvt_bt_kernel<<<dim3(VC_ / 32, D_ / 32), 256, 0, stream>>>(
        Wout, Wbf, D_, V_, c * VC_);
    gemm_bf16<<<dim3(VC_ / 128, LB_ / 128), 256, 0, stream>>>(
        Abf, Wbf, bout + c * VC_, Ubuf, D_, VC_);
    rowstat_kernel<<<dim3(LB_), 256, 0, stream>>>(
        Ubuf, VC_, c * VC_, y, sm, ss, st, c == 0);
  }
  loss_kernel<<<dim3(LB_ / 256), 256, 0, stream>>>(sm, ss, st, (float*)d_out);
}
